// WordKVMN_6725918785722
// MI455X (gfx1250) — compile-verified, matrix operand order checked by reading, not run on hardware
//
#include <hip/hip_runtime.h>

#define Bc 4
#define Lc 256
#define Wc 256
#define Hc 768

typedef __attribute__((ext_vector_type(16))) __bf16 v16bf;
typedef __attribute__((ext_vector_type(8)))  float  v8f;

union FragU {
    uint4 u[2];
    v16bf v;
};

__device__ __forceinline__ unsigned int f2bf(float f) {
    unsigned int x = __float_as_uint(f);
    // round-to-nearest-even to bf16
    return (x + 0x7FFFu + ((x >> 16) & 1u)) >> 16;
}

__device__ __forceinline__ unsigned int pack2bf(float a, float b) {
    return f2bf(a) | (f2bf(b) << 16);
}

// Kernel 1: convert hidden_state to bf16 and gather emb_a[word_seq] as bf16.
// Each thread handles 4 floats -> one uint2 (4 bf16).
__global__ void wkv_prep_kernel(const int* __restrict__ word_seq,
                                const float* __restrict__ hidden,
                                const float* __restrict__ emb_a,
                                uint2* __restrict__ hsb,   // [B*L*H/4] packed bf16
                                uint2* __restrict__ eab) { // [B*W*H/4] packed bf16
    const int n4 = Bc * Lc * Hc / 4; // 196608
    int tid = blockIdx.x * blockDim.x + threadIdx.x;
    if (tid < n4) {
        float4 v = ((const float4*)hidden)[tid];
        hsb[tid] = make_uint2(pack2bf(v.x, v.y), pack2bf(v.z, v.w));
    } else if (tid < 2 * n4) {
        int t  = tid - n4;
        int h4 = t % (Hc / 4);
        int bw = t / (Hc / 4);          // b*W + w
        int row = word_seq[bw];
        float4 v = ((const float4*)(emb_a + (size_t)row * Hc))[h4];
        eab[t] = make_uint2(pack2bf(v.x, v.y), pack2bf(v.z, v.w));
    }
}

// Kernel 2: fused bf16-WMMA attention scores + masked exp + label scatter +
// label-GEMM recombination.  One block per (batch, 16-row L tile).
// 512 threads = 16 waves; wave i computes the 16x16 u-tile at W columns [16i,16i+16).
__global__ __launch_bounds__(512)
void wkv_attn_kernel(const unsigned short* __restrict__ hsb,  // bf16 [B,L,H]
                     const unsigned short* __restrict__ eab,  // bf16 [B,W,H]
                     const int* __restrict__ labels,          // [B,L,W]
                     const float* __restrict__ mask,          // [B,L,W]
                     const float* __restrict__ emb_c,         // [10,H]
                     const float* __restrict__ hidden,        // f32 [B,L,H]
                     float* __restrict__ out) {               // f32 [B,L,H]
    __shared__ float embc_sh[10 * Hc];  // 30720 B
    __shared__ float Ssh[16][16];       // per-row label sums (10 used, padded)

    const int tid  = threadIdx.x;
    const int b    = blockIdx.x >> 4;
    const int l0   = (blockIdx.x & 15) << 4;
    const int lane = tid & 31;
    const int wv   = tid >> 5;          // wave id 0..15
    const int w0   = wv << 4;
    const int rsel = (lane >> 4) << 3;  // 0 for lanes 0-15, 8 for lanes 16-31
    const int rlow = lane & 15;

    // stage emb_c into LDS, zero S
    for (int i = tid; i < 10 * Hc; i += 512) embc_sh[i] = emb_c[i];
    if (tid < 256) ((float*)Ssh)[tid] = 0.0f;
    __syncthreads();

    // ---- GEMM1: u = hidden @ ea^T via v_wmma_f32_16x16x32_bf16 ----
    const unsigned short* arow = hsb + ((size_t)(b * Lc + l0 + rlow)) * Hc;
    const unsigned short* brow = eab + ((size_t)(b * Wc + w0 + rlow)) * Hc;
    v8f acc = {0.f, 0.f, 0.f, 0.f, 0.f, 0.f, 0.f, 0.f};
#pragma unroll
    for (int k0 = 0; k0 < Hc; k0 += 32) {
        int base = k0 + rsel;  // K offset for this lane-half
        const uint4* ap = (const uint4*)(arow + base);
        const uint4* bp = (const uint4*)(brow + base);
        FragU af, bf;
        af.u[0] = ap[0];  af.u[1] = ap[2];   // K: base..base+7, base+16..base+23
        bf.u[0] = bp[0];  bf.u[1] = bp[2];
        acc = __builtin_amdgcn_wmma_f32_16x16x32_bf16(
            false, af.v, false, bf.v, (short)0, acc, false, false);
    }

    // ---- masked exp + scatter-add into S[row][label] ----
    const float inv_temper = 0.03608439182435161f; // 1/sqrt(768)
#pragma unroll
    for (int r = 0; r < 8; ++r) {
        int lrow = r + rsel;                 // 0..15 within tile
        int l = l0 + lrow;
        int w = w0 + rlow;
        size_t idx = ((size_t)(b * Lc + l)) * Wc + w;
        float m = mask[idx];
        m = fminf(fmaxf(m, 0.0f), 1.0f);
        float e = __expf(acc[r] * inv_temper) * m;
        int k = labels[idx];
        atomicAdd(&Ssh[lrow][k], e);         // ds_add_f32
    }
    __syncthreads();

    // ---- output: hidden + (S @ emb_c) / (sum_k S + 1e-10), coalesced ----
#pragma unroll
    for (int c = 0; c < 24; ++c) {
        int idx = c * 512 + tid;             // 0..12287 over 16 rows x 768 cols
        int lrow = idx / Hc;
        int h = idx - lrow * Hc;
        float denom = 1e-10f;
        float o = 0.0f;
#pragma unroll
        for (int k = 0; k < 10; ++k) {
            float s = Ssh[lrow][k];
            denom += s;
            o += s * embc_sh[k * Hc + h];
        }
        size_t g = ((size_t)(b * Lc + l0 + lrow)) * Hc + h;
        out[g] = hidden[g] + o / denom;
    }
}

extern "C" void kernel_launch(void* const* d_in, const int* in_sizes, int n_in,
                              void* d_out, int out_size, void* d_ws, size_t ws_size,
                              hipStream_t stream) {
    const int*   word_seq = (const int*)d_in[0];
    const float* hidden   = (const float*)d_in[1];
    const int*   labels   = (const int*)d_in[2];
    const float* mask     = (const float*)d_in[3];
    const float* emb_a    = (const float*)d_in[4];
    const float* emb_c    = (const float*)d_in[5];
    float* out = (float*)d_out;

    // workspace: bf16 hidden [B*L*H] then bf16 gathered keys [B*W*H]
    unsigned short* hsb = (unsigned short*)d_ws;
    unsigned short* eab = hsb + (size_t)Bc * Lc * Hc;

    // prep: 2 * (B*L*H/4) = 393216 threads of 4 elements each
    wkv_prep_kernel<<<1536, 256, 0, stream>>>(word_seq, hidden, emb_a,
                                              (uint2*)hsb, (uint2*)eab);

    // fused attention: one block per (b, 16-row tile) = 64 blocks x 512 threads
    wkv_attn_kernel<<<Bc * (Lc / 16), 512, 0, stream>>>(hsb, eab, labels, mask,
                                                        emb_c, hidden, out);
}
